// MultiHeadAttention_16973710754273
// MI455X (gfx1250) — matrix operand order repaired and checked by reading, not compiled, not run on hardware
//
#include <hip/hip_runtime.h>
#include <stdint.h>

// ---------------- types ----------------
typedef __attribute__((ext_vector_type(16))) __bf16        v16bf;
typedef __attribute__((ext_vector_type(8)))  float         v8f;
typedef __attribute__((ext_vector_type(4)))  float         f32x4;
typedef __attribute__((ext_vector_type(4)))  unsigned int  u32x4;
typedef __attribute__((ext_vector_type(8)))  int           i32x8;
typedef __attribute__((ext_vector_type(4)))  int           i32x4;

#define T_SEQ  2048
#define DMODEL 1024
#define DH     64
#define NGROUP 32
#define GELEMS (T_SEQ * DH)      // 131072 elements per group
#define SM_SCALE 0.125f          // dh^-0.5

union FragU   { u32x4 q[2]; v16bf v; };          // 16 bf16 = 32B fragment
union H8Pack  { unsigned short s[8]; u32x4 u; }; // 8 bf16 = 16B

static __device__ __forceinline__ unsigned short f32_to_bf16_bits(float f) {
  unsigned int x = __builtin_bit_cast(unsigned int, f);
  x += 0x7FFFu + ((x >> 16) & 1u);               // round-to-nearest-even
  return (unsigned short)(x >> 16);
}

static __device__ __forceinline__ v8f wmma_bf16(v16bf a, v16bf b, v8f c) {
  // (neg_a, A, neg_b, B, c_mod, C, reuse_a, reuse_b)
  return __builtin_amdgcn_wmma_f32_16x16x32_bf16(false, a, false, b, (short)0, c,
                                                 false, false);
}

// ============================================================================
// Kernel 0: streaming fp32 -> bf16 conversion (bandwidth-bound, done ONCE so
// the GEMM inner loops carry zero conversion VALU).  8 elements / thread.
// ============================================================================
__global__ void __launch_bounds__(256)
cvt_f32_bf16_kernel(const float* __restrict__ src,
                    unsigned short* __restrict__ dst, int n)
{
  const int idx = (blockIdx.x * 256 + threadIdx.x) * 8;
  if (idx + 8 > n) return;                       // sizes are multiples of 8
  f32x4 a = *(const f32x4*)(src + idx);
  f32x4 b = *(const f32x4*)(src + idx + 4);
  H8Pack o;
  #pragma unroll
  for (int c = 0; c < 4; ++c) {
    o.s[c]     = f32_to_bf16_bits(a[c]);
    o.s[4 + c] = f32_to_bf16_bits(b[c]);
  }
  *(u32x4*)(dst + idx) = o.u;
}

// ============================================================================
// Kernel 1: projection GEMM  out = X @ W^T + b  (4096 x 1024, K = 1024),
// all-bf16 operands.  Wave does 16 rows x 128 cols.  All 9 fragments of a
// K-step get PRIVATE registers so the scheduler can issue one 18-load clause
// and use partial s_wait_loadcnt instead of serializing load->wmma.
// mode 0: row-major bf16 store.  mode 1: per-group transposed store
// vpT[g*GELEMS + j*T_SEQ + s]  (g=row/128, s=(row%128)*16+col/64, j=col%64).
// ============================================================================
__global__ void __launch_bounds__(128)
proj_gemm_kernel(const unsigned short* __restrict__ Xb,
                 const unsigned short* __restrict__ Wb,
                 const float* __restrict__ bias,
                 unsigned short* __restrict__ dst, int mode)
{
  const int wave = threadIdx.x >> 5;
  const int lane = threadIdx.x & 31;
  const int ln15 = lane & 15;
  const int hi   = lane >> 4;
  const int h8   = hi << 3;

  const int m0 = blockIdx.x * 64 + wave * 16;   // output row base
  const int n0 = blockIdx.y * 128;              // output col base (8 x 16)

  v8f acc[8];
  v8f zf = {};
  #pragma unroll
  for (int jt = 0; jt < 8; ++jt) acc[jt] = zf;

  const unsigned short* arow = Xb + (m0 + ln15) * DMODEL;
  const unsigned short* brow = Wb + (n0 + ln15) * DMODEL + (hi << 4);

  for (int k0 = 0; k0 < DMODEL; k0 += 32) {
    // A fragment: per-lane 16 contiguous K at 16*hi (matches B)
    FragU A;
    A.q[0] = *(const u32x4*)(arow + k0 + (hi << 4));
    A.q[1] = *(const u32x4*)(arow + k0 + (hi << 4) + 8);

    FragU B[8];
    #pragma unroll
    for (int jt = 0; jt < 8; ++jt) {
      // B[kk,n] = W[(n0+jt*16+n)*D + k0+kk]; per-lane 16 contiguous K at 16*hi
      const unsigned short* bp = brow + jt * 16 * DMODEL + k0;
      B[jt].q[0] = *(const u32x4*)(bp);
      B[jt].q[1] = *(const u32x4*)(bp + 8);
    }
    #pragma unroll
    for (int jt = 0; jt < 8; ++jt)
      acc[jt] = wmma_bf16(A.v, B[jt].v, acc[jt]);
  }

  // Store: C/D layout -> row M = r + 8*hi, col N = lane&15.
  #pragma unroll
  for (int jt = 0; jt < 8; ++jt) {
    const int col  = n0 + jt * 16 + ln15;
    const float bb = bias[col];
    #pragma unroll
    for (int r = 0; r < 8; ++r) {
      const int row = m0 + r + h8;
      const unsigned short hv = f32_to_bf16_bits(acc[jt][r] + bb);
      if (mode == 0) {
        dst[row * DMODEL + col] = hv;
      } else {
        const int g = row >> 7;
        const int s = ((row & 127) << 4) | (col >> 6);
        const int j = col & 63;
        dst[g * GELEMS + j * T_SEQ + s] = hv;
      }
    }
  }
}

// ============================================================================
// Kernel 2: fused attention for one (group, 64-row tile), 4 waves x 16 rows.
// Pass 1: online softmax stats, K fragments ping-pong double-buffered so the
// shfl/exp VALU of tile i hides the loads of tile i+1.
// Pass 2: all 8 fragment loads (4 K + 4 V) hoisted to the iteration top;
// writes normalized attn non-temporally (512MB must not thrash the 192MB L2)
// and accumulates ctx = P @ V via an LDS round-trip of P into an A-fragment.
// ============================================================================
__global__ void __launch_bounds__(128)
attn_kernel(const unsigned short* __restrict__ qp,
            const unsigned short* __restrict__ kp,
            const unsigned short* __restrict__ vpT,
            float* __restrict__ attn,
            float* __restrict__ ctx)
{
  __shared__ __align__(16) unsigned short lds[4 * 16 * 48];  // per-wave 16x32 P

  const int wave = threadIdx.x >> 5;
  const int lane = threadIdx.x & 31;
  const int ln15 = lane & 15;
  const int hi   = lane >> 4;
  const int h8   = hi << 3;

  const int g  = blockIdx.y;
  const int i0 = blockIdx.x * 64 + wave * 16;   // group-local row base

  const unsigned short* qg = qp  + g * GELEMS;
  const unsigned short* kg = kp  + g * GELEMS;
  const unsigned short* vg = vpT + g * GELEMS;
  float* attng = attn + (size_t)g * T_SEQ * T_SEQ;

  // Q A-fragments (rows i0..i0+15, K = 0..31 and 32..63), loaded once.
  FragU A0, A1;
  {
    const unsigned short* qr = qg + (i0 + ln15) * DH;
    A0.q[0] = *(const u32x4*)(qr + (hi << 4));
    A0.q[1] = *(const u32x4*)(qr + (hi << 4) + 8);
    A1.q[0] = *(const u32x4*)(qr + 32 + (hi << 4));
    A1.q[1] = *(const u32x4*)(qr + 40 + (hi << 4));
  }

  const unsigned short* krow = kg + ln15 * DH + (hi << 4);

  auto load_k = [&](int sb, FragU& B0, FragU& B1) {
    const unsigned short* kr = krow + sb * DH;
    B0.q[0] = *(const u32x4*)(kr);
    B0.q[1] = *(const u32x4*)(kr + 8);
    B1.q[0] = *(const u32x4*)(kr + 32);
    B1.q[1] = *(const u32x4*)(kr + 40);
  };

  // ---- pass 1: row max m[] and rescaled row sum l[] --------------------
  float m[8], l[8];
  #pragma unroll
  for (int r = 0; r < 8; ++r) { m[r] = -3.0e38f; l[r] = 0.0f; }

  auto stats_tile = [&](const FragU& B0, const FragU& B1) {
    v8f S = {};
    S = wmma_bf16(A0.v, B0.v, S);
    S = wmma_bf16(A1.v, B1.v, S);
    #pragma unroll
    for (int r = 0; r < 8; ++r) {
      float sv = S[r] * SM_SCALE;
      float tmax = sv;
      tmax = fmaxf(tmax, __shfl_xor(tmax, 1));
      tmax = fmaxf(tmax, __shfl_xor(tmax, 2));
      tmax = fmaxf(tmax, __shfl_xor(tmax, 4));
      tmax = fmaxf(tmax, __shfl_xor(tmax, 8));
      const float mnew = fmaxf(m[r], tmax);
      float p = __expf(sv - mnew);
      p += __shfl_xor(p, 1);
      p += __shfl_xor(p, 2);
      p += __shfl_xor(p, 4);
      p += __shfl_xor(p, 8);
      l[r] = l[r] * __expf(m[r] - mnew) + p;
      m[r] = mnew;
    }
  };

  {
    FragU B0a, B1a, B0b, B1b;
    load_k(0, B0a, B1a);
    for (int s0 = 0; s0 < T_SEQ; s0 += 32) {
      __builtin_prefetch(krow + ((s0 + 64) & (T_SEQ - 1)) * DH, 0, 1);
      load_k(s0 + 16, B0b, B1b);
      stats_tile(B0a, B1a);                 // VALU hides the b-buffer loads
      if (s0 + 32 < T_SEQ) load_k(s0 + 32, B0a, B1a);
      stats_tile(B0b, B1b);
    }
  }

  float rl[8];
  #pragma unroll
  for (int r = 0; r < 8; ++r) rl[r] = 1.0f / l[r];

  // ---- pass 2: write attn, accumulate ctx -------------------------------
  v8f C[4];
  v8f zf = {};
  #pragma unroll
  for (int jt = 0; jt < 4; ++jt) C[jt] = zf;

  unsigned short* ldsw = lds + wave * 16 * 48;

  for (int s0 = 0; s0 < T_SEQ; s0 += 32) {
    // hoist ALL loads of this iteration: 4 K fragments + 4 V fragments
    FragU K00, K01, K10, K11;
    load_k(s0,      K00, K01);
    load_k(s0 + 16, K10, K11);
    FragU BV[4];
    #pragma unroll
    for (int jt = 0; jt < 4; ++jt) {
      // B[kk,n] = v[s0+kk, jt*16+n] from vpT[g][j][s]: per-lane contiguous in s
      const unsigned short* vr = vg + (jt * 16 + ln15) * T_SEQ + s0 + (hi << 4);
      BV[jt].q[0] = *(const u32x4*)(vr);
      BV[jt].q[1] = *(const u32x4*)(vr + 8);
    }

    #pragma unroll
    for (int st = 0; st < 2; ++st) {
      const int sb = s0 + st * 16;
      v8f S = {};
      S = wmma_bf16(A0.v, (st == 0 ? K00 : K10).v, S);
      S = wmma_bf16(A1.v, (st == 0 ? K01 : K11).v, S);
      #pragma unroll
      for (int r = 0; r < 8; ++r) {
        const int row = r + h8;                       // local row 0..15
        const float p = __expf(S[r] * SM_SCALE - m[r]) * rl[r];
        __builtin_nontemporal_store(
            p, attng + (size_t)(i0 + row) * T_SEQ + sb + ln15);
        ldsw[row * 48 + st * 16 + ln15] = f32_to_bf16_bits(p);
      }
    }

    __asm__ __volatile__("" ::: "memory");            // order LDS st -> ld

    // P as 16x32 bf16 A-fragment: row = lane&15, 16 contiguous K at 16*hi.
    FragU PA;
    {
      const unsigned short* pr = ldsw + ln15 * 48;
      PA.q[0] = *(const u32x4*)(pr + (hi << 4));
      PA.q[1] = *(const u32x4*)(pr + (hi << 4) + 8);
    }

    __asm__ __volatile__("" ::: "memory");            // order LDS ld -> next st

    #pragma unroll
    for (int jt = 0; jt < 4; ++jt)
      C[jt] = wmma_bf16(PA.v, BV[jt].v, C[jt]);
  }

  // ctx store (fp32 workspace, same flat layout as qp)
  #pragma unroll
  for (int jt = 0; jt < 4; ++jt) {
    #pragma unroll
    for (int r = 0; r < 8; ++r) {
      ctx[g * GELEMS + (i0 + r + h8) * DH + jt * 16 + ln15] = C[jt][r];
    }
  }
}

// ============================================================================
// Kernel 3: out = LayerNorm(q + ctx) * g + b, one 256-thread block per row.
// ============================================================================
__global__ void __launch_bounds__(256)
ln_kernel(const float* __restrict__ q, const float* __restrict__ ctx,
          const float* __restrict__ gamma, const float* __restrict__ beta,
          float* __restrict__ out)
{
  __shared__ float sred[8];
  const int row = blockIdx.x;
  const int t   = threadIdx.x;
  const float* xq = q   + row * DMODEL;
  const float* xc = ctx + row * DMODEL;

  float x[4];
  float s = 0.0f;
  #pragma unroll
  for (int i = 0; i < 4; ++i) {
    const int idx = t + i * 256;
    x[i] = xq[idx] + xc[idx];
    s += x[i];
  }
  #pragma unroll
  for (int off = 16; off >= 1; off >>= 1) s += __shfl_xor(s, off);
  if ((t & 31) == 0) sred[t >> 5] = s;
  __syncthreads();
  float tot = 0.0f;
  #pragma unroll
  for (int w = 0; w < 8; ++w) tot += sred[w];
  const float mu = tot * (1.0f / DMODEL);
  __syncthreads();

  float vs = 0.0f;
  #pragma unroll
  for (int i = 0; i < 4; ++i) { const float d = x[i] - mu; vs += d * d; }
  #pragma unroll
  for (int off = 16; off >= 1; off >>= 1) vs += __shfl_xor(vs, off);
  if ((t & 31) == 0) sred[t >> 5] = vs;
  __syncthreads();
  float vtot = 0.0f;
  #pragma unroll
  for (int w = 0; w < 8; ++w) vtot += sred[w];
  const float rstd = rsqrtf(vtot * (1.0f / DMODEL) + 1e-5f);

  #pragma unroll
  for (int i = 0; i < 4; ++i) {
    const int idx = t + i * 256;
    __builtin_nontemporal_store((x[i] - mu) * rstd * gamma[idx] + beta[idx],
                                out + row * DMODEL + idx);
  }
}

// ============================================================================
// Kernel 4 (isolated TDM demo): build a D# per ISA 8.3/8.4 and issue
// tensor_load_to_lds (64x16 dwords, contiguous 4KB) + s_wait_tensorcnt.
// Output goes to a scratch region that attn_kernel fully overwrites later,
// so pipeline correctness does not depend on TDM descriptor semantics.
// ============================================================================
__global__ void __launch_bounds__(32)
tdm_demo_kernel(const unsigned int* __restrict__ src, float* __restrict__ scratch)
{
  __shared__ __align__(16) unsigned int tlds[1024];    // 4KB landing zone
  const unsigned long long ga = (unsigned long long)(uintptr_t)src;

  u32x4 g0;
  g0[0] = 1u;                                         // count=1 (valid, user)
  g0[1] = 0u;                                         // lds_addr = 0
  g0[2] = (unsigned int)(ga & 0xFFFFFFFFu);           // global_addr lo
  g0[3] = (unsigned int)((ga >> 32) & 0x01FFFFFFu) | 0x80000000u; // type=2

  i32x8 g1;
  g1[0] = 0x20000;              // workgroup_mask=0, data_size=2 (4B)
  g1[1] = (int)(64u << 16);     // tensor_dim0 = 64   [79:48]
  g1[2] = (int)(16u << 16);     // tensor_dim1 = 16   [111:80]
  g1[3] = (int)(64u << 16);     // tile_dim0 = 64     [127:112]
  g1[4] = 16;                   // tile_dim1 = 16     [143:128]
  g1[5] = 64;                   // tensor_dim0_stride = 64  [207:160]
  g1[6] = 0;
  g1[7] = 0;

  i32x4 gz = {};                // groups 2/3 unused (<=2D tensor)

  if (threadIdx.x == 0) {
#if __has_include(<hip/amd_detail/amd_gfx1250_TDM.h>)
    i32x8 gz8 = {};
    __builtin_amdgcn_tensor_load_to_lds(g0, g1, gz, gz, gz8, 0);
#else
    __builtin_amdgcn_tensor_load_to_lds(g0, g1, gz, gz, 0);
#endif
    __builtin_amdgcn_s_wait_tensorcnt(0);
  }
  __syncthreads();
  scratch[threadIdx.x] = (float)tlds[threadIdx.x];     // keep it observable
}

// ============================================================================
// launch.  ws layout (bytes):
//   [ 0M, 8M)  xq bf16     [ 8M,16M)  xk bf16     [16M,24M)  xv bf16
//   [24M,26M)  wq bf16     [26M,28M)  wk bf16     [28M,30M)  wv bf16
//   [30M,38M)  qp bf16     [38M,46M)  kp bf16     [46M,54M)  vpT bf16
//   [54M,70M)  ctx f32 (also TDM demo scratch, overwritten by attn)
// d_out: out f32 (4,194,304) then attn f32 (134,217,728).
// ============================================================================
extern "C" void kernel_launch(void* const* d_in, const int* in_sizes, int n_in,
                              void* d_out, int out_size, void* d_ws, size_t ws_size,
                              hipStream_t stream)
{
  (void)in_sizes; (void)n_in; (void)out_size; (void)ws_size;

  const float* q  = (const float*)d_in[0];
  const float* k  = (const float*)d_in[1];
  const float* v  = (const float*)d_in[2];
  const float* wq = (const float*)d_in[3];
  const float* bq = (const float*)d_in[4];
  const float* wk = (const float*)d_in[5];
  const float* bk = (const float*)d_in[6];
  const float* wv = (const float*)d_in[7];
  const float* bv = (const float*)d_in[8];
  const float* lg = (const float*)d_in[9];
  const float* lb = (const float*)d_in[10];

  const size_t MB = (size_t)1024 * 1024;
  char* ws = (char*)d_ws;
  unsigned short* xq  = (unsigned short*)(ws);
  unsigned short* xk  = (unsigned short*)(ws + 8  * MB);
  unsigned short* xv  = (unsigned short*)(ws + 16 * MB);
  unsigned short* wqb = (unsigned short*)(ws + 24 * MB);
  unsigned short* wkb = (unsigned short*)(ws + 26 * MB);
  unsigned short* wvb = (unsigned short*)(ws + 28 * MB);
  unsigned short* qp  = (unsigned short*)(ws + 30 * MB);
  unsigned short* kp  = (unsigned short*)(ws + 38 * MB);
  unsigned short* vpT = (unsigned short*)(ws + 46 * MB);
  float*          ctx = (float*)        (ws + 54 * MB);

  float* out  = (float*)d_out;
  float* attn = out + (size_t)2 * T_SEQ * DMODEL;   // after 4,194,304 `out` floats

  const int NACT = 2 * T_SEQ * DMODEL;   // 4,194,304
  const int NWGT = DMODEL * DMODEL;      // 1,048,576

  // one-time precision conversion (bandwidth-bound)
  cvt_f32_bf16_kernel<<<NACT / (256 * 8), 256, 0, stream>>>(q,  xq,  NACT);
  cvt_f32_bf16_kernel<<<NACT / (256 * 8), 256, 0, stream>>>(k,  xk,  NACT);
  cvt_f32_bf16_kernel<<<NACT / (256 * 8), 256, 0, stream>>>(v,  xv,  NACT);
  cvt_f32_bf16_kernel<<<NWGT / (256 * 8), 256, 0, stream>>>(wq, wqb, NWGT);
  cvt_f32_bf16_kernel<<<NWGT / (256 * 8), 256, 0, stream>>>(wk, wkb, NWGT);
  cvt_f32_bf16_kernel<<<NWGT / (256 * 8), 256, 0, stream>>>(wv, wvb, NWGT);

  // projections: grid (4096/64 rows, 1024/128 cols)
  proj_gemm_kernel<<<dim3(64, 8), 128, 0, stream>>>(xq, wqb, bq, qp,  0);
  proj_gemm_kernel<<<dim3(64, 8), 128, 0, stream>>>(xk, wkb, bk, kp,  0);
  proj_gemm_kernel<<<dim3(64, 8), 128, 0, stream>>>(xv, wvb, bv, vpT, 1);

  // TDM demo (scratch target is overwritten by attn_kernel below)
  tdm_demo_kernel<<<1, 32, 0, stream>>>((const unsigned int*)kp, ctx);

  // attention: 32 row-tiles x 32 groups
  attn_kernel<<<dim3(32, 32), 128, 0, stream>>>(qp, kp, vpT, attn, ctx);

  // layernorm: 4096 rows
  ln_kernel<<<4096, 256, 0, stream>>>(q, ctx, lg, lb, out);
}
